// MultiHeadAttention_72773925864105
// MI455X (gfx1250) — matrix operand order repaired and checked by reading, not compiled, not run on hardware
//
#include <hip/hip_runtime.h>
#include <hip/hip_bf16.h>

// MHA forward for gfx1250 (MI455X), bf16 WMMA + fp32 accumulate.
// Attention stages K/V tiles into LDS via the Tensor Data Mover (TDM),
// double-buffered and shared by all 8 waves of a workgroup.
//
// Workspace layout (bytes):
//   [0)        Xb   : bf16 x            (8192x1024)          16 MB
//   [16MB)     WqT  : bf16 W_qkv^T      (3072x1024, k-contig)  6 MB
//   [22MB)     WoT  : bf16 W_out^T      (1024x1024, k-contig)  2 MB
//   [24MB)     Q    : bf16 (B,H,T,hd), pre-scaled by 1/sqrt(hd) 16 MB
//   [40MB)     K    : bf16 (B,H,T,hd)                          16 MB
//   [56MB)     Vt   : bf16 (B,H,hd,T)  (key-contiguous)        16 MB
//   [72MB)     Ob   : bf16 attention out (B,T,D)               16 MB

#define DMODEL 1024
#define NHEADS 16
#define HDIM   64
#define BATCH  4
#define SEQ    2048
#define BT     (BATCH*SEQ)     // 8192
#define NQKV   (3*DMODEL)      // 3072

typedef __attribute__((ext_vector_type(16))) __bf16 v16bf;
typedef __attribute__((ext_vector_type(8)))  float  v8f;
typedef __attribute__((ext_vector_type(4)))  unsigned tdm_v4u;
typedef __attribute__((ext_vector_type(8)))  int      tdm_v8i;
typedef __attribute__((ext_vector_type(4)))  int      tdm_v4i;

// Load a 16x32 A-frag half-pair / 32x16 B-frag as two 16B chunks (global or LDS).
__device__ __forceinline__ v16bf frag_ld(const __bf16* p0, const __bf16* p1) {
  union { uint4 u[2]; v16bf v; } f;
  f.u[0] = *(const uint4*)p0;
  f.u[1] = *(const uint4*)p1;
  return f.v;
}
__device__ __forceinline__ v8f vzero8() {
  v8f z = {0.f,0.f,0.f,0.f,0.f,0.f,0.f,0.f};
  return z;
}
__device__ __forceinline__ v8f wmma_bf16(v16bf a, v16bf b, v8f c) {
  return __builtin_amdgcn_wmma_f32_16x16x32_bf16(false, a, false, b, (short)0, c, false, false);
}

// ---- TDM: 2-D tile load (bf16 elements) per CDNA5 ISA ----
// D# group0: [1:0]=count=1, [63:32]=lds_addr, [120:64]=global_addr, [127:126]=type=2
// D# group1: [17:16]=data_size(1 => 2 bytes), [79:48]=tensor_dim0, [111:80]=tensor_dim1,
//            [127:112]=tile_dim0, [143:128]=tile_dim1, [207:160]=tensor_dim0_stride
// groups 2/3: zero (tensor is 2-D). Dims/strides are in element (data_size) units.
// This toolchain's builtin is the 6-arg form:
//   (uint32x4 g0, int32x8 g1, int32x4 g2, int32x4 g3, int32x8 pad, i32 cpol)
__device__ __forceinline__ void tdm_load_2d(const void* gaddr, unsigned lds_addr,
                                            unsigned tensor_d0, unsigned tensor_d1,
                                            unsigned long long stride0,
                                            unsigned tile_d0, unsigned tile_d1) {
  unsigned long long ga = (unsigned long long)gaddr;
  tdm_v4u g0;
  g0[0] = 1u;                                           // count=1 (user descriptor)
  g0[1] = lds_addr;                                     // LDS byte address
  g0[2] = (unsigned)(ga & 0xffffffffu);                 // global_addr[31:0]
  g0[3] = (unsigned)((ga >> 32) & 0x1ffffffu) | (2u << 30); // addr[56:32] | type=2
  tdm_v8i g1;
  g1[0] = (int)(1u << 16);                              // data_size = 2 bytes
  g1[1] = (int)((tensor_d0 & 0xffffu) << 16);           // tensor_dim0[15:0]
  g1[2] = (int)(((tensor_d0 >> 16) & 0xffffu) | ((tensor_d1 & 0xffffu) << 16));
  g1[3] = (int)(((tensor_d1 >> 16) & 0xffffu) | ((tile_d0 & 0xffffu) << 16));
  g1[4] = (int)(tile_d1 & 0xffffu);                     // tile_dim1, tile_dim2=0
  g1[5] = (int)(unsigned)(stride0 & 0xffffffffu);       // stride0[31:0]
  g1[6] = (int)(unsigned)((stride0 >> 32) & 0xffffu);   // stride0[47:32]
  g1[7] = 0;
  tdm_v4i z4 = {0, 0, 0, 0};
  tdm_v8i z8 = {0, 0, 0, 0, 0, 0, 0, 0};
  __builtin_amdgcn_tensor_load_to_lds(g0, g1, z4, z4, z8, 0);
}
// Generic shared pointer -> LDS byte offset (flat LDS aperture keeps offset in [31:0]).
__device__ __forceinline__ unsigned lds_off(const void* p) {
  return (unsigned)(unsigned long long)p;
}

// ---------------- conversion kernels ----------------
__global__ void cvt_f32_bf16(const float* __restrict__ src, __bf16* __restrict__ dst, int n) {
  int i = blockIdx.x * blockDim.x + threadIdx.x;
  if (i < n) dst[i] = (__bf16)src[i];
}

// W: (Din x Dout) fp32 row-major  ->  Wt: (Dout x Din) bf16 row-major (k contiguous)
__global__ void cvt_transpose(const float* __restrict__ W, __bf16* __restrict__ Wt,
                              int Din, int Dout) {
  int i = blockIdx.x * blockDim.x + threadIdx.x;
  if (i >= Din * Dout) return;
  int d = i % Din;
  int e = i / Din;
  Wt[i] = (__bf16)W[d * Dout + e];
}

// ---------------- QKV projection GEMM ----------------
__global__ __launch_bounds__(256) void qkv_gemm(const __bf16* __restrict__ Xb,
                                                const __bf16* __restrict__ WqT,
                                                const float*  __restrict__ bqkv,
                                                __bf16* __restrict__ Q,
                                                __bf16* __restrict__ Kd,
                                                __bf16* __restrict__ Vt) {
  const int lane = threadIdx.x & 31;
  const int wave = (blockIdx.x * blockDim.x + threadIdx.x) >> 5;
  const int NT = NQKV / 64;                 // 48 column tiles
  const int nt = wave % NT;
  const int mt = wave / NT;                 // 0..511
  const int m  = lane & 15;
  const int hf = lane >> 4;

  const __bf16* arow = Xb + (size_t)(mt * 16 + m) * DMODEL;
  const __bf16* brow[4];
#pragma unroll
  for (int s = 0; s < 4; ++s)
    brow[s] = WqT + (size_t)(nt * 64 + s * 16 + m) * DMODEL;

  v8f acc[4];
#pragma unroll
  for (int s = 0; s < 4; ++s) acc[s] = vzero8();

  for (int k0 = 0; k0 < DMODEL; k0 += 32) {
    __builtin_prefetch(arow + k0 + 256, 0, 1);
    v16bf a = frag_ld(arow + k0 + hf * 8, arow + k0 + 16 + hf * 8);
#pragma unroll
    for (int s = 0; s < 4; ++s) {
      v16bf b = frag_ld(brow[s] + k0 + hf * 8, brow[s] + k0 + 16 + hf * 8);
      acc[s] = wmma_bf16(a, b, acc[s]);
    }
  }

#pragma unroll
  for (int s = 0; s < 4; ++s) {
    const int e     = nt * 64 + s * 16 + m;   // C-layout column = lane%16
    const int which = e >> 10;                // 0=Q 1=K 2=V
    const int h     = (e & 1023) >> 6;
    const int d     = e & 63;
    const float bias = bqkv[e];
#pragma unroll
    for (int r = 0; r < 8; ++r) {
      const int t  = mt * 16 + hf * 8 + r;    // global row (b*T + t)
      const int b  = t >> 11;
      const int tl = t & 2047;
      const float val = acc[s][r] + bias;
      if (which == 0) {
        Q[(((size_t)(b * NHEADS + h) * SEQ) + tl) * HDIM + d] = (__bf16)(val * 0.125f);
      } else if (which == 1) {
        Kd[(((size_t)(b * NHEADS + h) * SEQ) + tl) * HDIM + d] = (__bf16)val;
      } else {
        Vt[(((size_t)(b * NHEADS + h) * HDIM) + d) * SEQ + tl] = (__bf16)val;
      }
    }
  }
}

// ---------------- flash attention (TDM-staged K/V tiles) ----------------
// Block = 8 waves = 8 query tiles of the SAME (b,h). Wave 0 drives the TDM to
// double-buffer 32-key K/V tiles into LDS; all waves consume them via ds loads.
__global__ __launch_bounds__(256) void attn_fwd(const __bf16* __restrict__ Q,
                                                const __bf16* __restrict__ Kd,
                                                const __bf16* __restrict__ Vt,
                                                __bf16* __restrict__ O) {
  __shared__ __bf16 Ks_s[2][32 * HDIM];      // 32 keys x 64 hd (hd contiguous)   4 KB ea
  __shared__ __bf16 Vs_s[2][HDIM * 32];      // 64 hd rows x 32 keys (key contig) 4 KB ea
  __shared__ __bf16 Pbuf[8][16 * 32];        // per-wave P relayout tile (1 KB each)

  const int lane = threadIdx.x & 31;
  const int wv   = threadIdx.x >> 5;
  const int bh   = blockIdx.x >> 4;          // 0..63, shared by whole block
  const int qt   = (blockIdx.x & 15) * 8 + wv; // query tile in [0,128)
  const int m  = lane & 15;
  const int hf = lane >> 4;

  const __bf16* Qp = Q  + (size_t)bh * SEQ * HDIM;
  const __bf16* Kp = Kd + (size_t)bh * SEQ * HDIM;
  const __bf16* Vp = Vt + (size_t)bh * HDIM * SEQ;
  __bf16* Pw = &Pbuf[wv][0];

  const __bf16* qr = Qp + (size_t)(qt * 16 + m) * HDIM;
  const v16bf qf0 = frag_ld(qr + hf * 8,      qr + 16 + hf * 8);   // hd 0..31
  const v16bf qf1 = frag_ld(qr + 32 + hf * 8, qr + 48 + hf * 8);   // hd 32..63

  float mrun[8], lrun[8];
  v8f acc[4];
#pragma unroll
  for (int r = 0; r < 8; ++r) { mrun[r] = -1e30f; lrun[r] = 0.f; }
#pragma unroll
  for (int s = 0; s < 4; ++s) acc[s] = vzero8();

  // Preload first K/V tile.
  if (threadIdx.x == 0) {
    tdm_load_2d(Kp, lds_off(&Ks_s[0][0]), HDIM, SEQ, HDIM, HDIM, 32);
    tdm_load_2d(Vp, lds_off(&Vs_s[0][0]), SEQ, HDIM, SEQ, 32, HDIM);
  }

  for (int kb = 0; kb < SEQ; kb += 32) {
    const int buf = (kb >> 5) & 1;
    if (threadIdx.x == 0) __builtin_amdgcn_s_wait_tensorcnt(0); // tile(kb) landed
    __syncthreads();                                            // visible to all waves
    if (threadIdx.x == 0 && kb + 32 < SEQ) {                    // prefetch next tile
      tdm_load_2d(Kp + (size_t)(kb + 32) * HDIM, lds_off(&Ks_s[buf ^ 1][0]),
                  HDIM, SEQ, HDIM, HDIM, 32);
      tdm_load_2d(Vp + (kb + 32), lds_off(&Vs_s[buf ^ 1][0]),
                  SEQ, HDIM, SEQ, 32, HDIM);
    }
    const __bf16* Kt = &Ks_s[buf][0];
    const __bf16* Vs = &Vs_s[buf][0];

    // ---- logits for two 16-key tiles (Q pre-scaled by 1/sqrt(hd)) ----
    v8f c0, c1;
#pragma unroll
    for (int kt = 0; kt < 2; ++kt) {
      const __bf16* kr = Kt + (kt * 16 + m) * HDIM;
      v16bf kf0 = frag_ld(kr + hf * 8,      kr + 16 + hf * 8);
      v16bf kf1 = frag_ld(kr + 32 + hf * 8, kr + 48 + hf * 8);
      v8f c = wmma_bf16(qf0, kf0, vzero8());
      c = wmma_bf16(qf1, kf1, c);
      if (kt == 0) c0 = c; else c1 = c;
    }

    // ---- online softmax: chunk row max, rescale, exp, row sum ----
    float tm[8], alpha[8], rs[8];
#pragma unroll
    for (int r = 0; r < 8; ++r) tm[r] = fmaxf(c0[r], c1[r]);
#pragma unroll
    for (int msk = 1; msk < 16; msk <<= 1)
#pragma unroll
      for (int r = 0; r < 8; ++r) tm[r] = fmaxf(tm[r], __shfl_xor(tm[r], msk, 32));
#pragma unroll
    for (int r = 0; r < 8; ++r) {
      float mn = fmaxf(mrun[r], tm[r]);
      alpha[r] = __expf(mrun[r] - mn);
      mrun[r] = mn;
      lrun[r] *= alpha[r];
    }
#pragma unroll
    for (int s = 0; s < 4; ++s)
#pragma unroll
      for (int r = 0; r < 8; ++r) acc[s][r] *= alpha[r];
#pragma unroll
    for (int r = 0; r < 8; ++r) {
      float p0 = __expf(c0[r] - mrun[r]);
      float p1 = __expf(c1[r] - mrun[r]);
      rs[r] = p0 + p1;
      const int rw = hf * 8 + r;             // C-layout row
      Pw[rw * 32 + m]      = (__bf16)p0;
      Pw[rw * 32 + 16 + m] = (__bf16)p1;
    }
#pragma unroll
    for (int msk = 1; msk < 16; msk <<= 1)
#pragma unroll
      for (int r = 0; r < 8; ++r) rs[r] += __shfl_xor(rs[r], msk, 32);
#pragma unroll
    for (int r = 0; r < 8; ++r) lrun[r] += rs[r];

    // ---- C-layout -> A-layout relayout via LDS (same-wave, in order) ----
    v16bf pf;
    {
      union { uint4 u[2]; v16bf v; } f;
      f.u[0] = *(const uint4*)&Pw[m * 32 + hf * 8];
      f.u[1] = *(const uint4*)&Pw[m * 32 + 16 + hf * 8];
      pf = f.v;
    }

    // ---- P @ V accumulate (V tile is key-contiguous in LDS) ----
#pragma unroll
    for (int s = 0; s < 4; ++s) {
      const __bf16* vr = Vs + (s * 16 + m) * 32 + hf * 8;
      v16bf vf = frag_ld(vr, vr + 16);
      acc[s] = wmma_bf16(pf, vf, acc[s]);
    }
    __syncthreads();   // all waves done with buf before TDM overwrites it
  }

  // ---- epilogue: normalize and store to (B,T,D) bf16 ----
  const int b = bh >> 4;
  const int h = bh & 15;
#pragma unroll
  for (int s = 0; s < 4; ++s) {
#pragma unroll
    for (int r = 0; r < 8; ++r) {
      const int t = qt * 16 + hf * 8 + r;
      const int dcol = h * HDIM + s * 16 + m;
      O[((size_t)(b * SEQ + t)) * DMODEL + dcol] = (__bf16)(acc[s][r] / lrun[r]);
    }
  }
}

// ---------------- output projection GEMM ----------------
__global__ __launch_bounds__(256) void out_gemm(const __bf16* __restrict__ Ob,
                                                const __bf16* __restrict__ WoT,
                                                const float*  __restrict__ bout,
                                                float* __restrict__ out) {
  const int lane = threadIdx.x & 31;
  const int wave = (blockIdx.x * blockDim.x + threadIdx.x) >> 5;
  const int NT = DMODEL / 64;               // 16 column tiles
  const int nt = wave % NT;
  const int mt = wave / NT;                 // 0..511
  const int m  = lane & 15;
  const int hf = lane >> 4;

  const __bf16* arow = Ob + (size_t)(mt * 16 + m) * DMODEL;
  const __bf16* brow[4];
#pragma unroll
  for (int s = 0; s < 4; ++s)
    brow[s] = WoT + (size_t)(nt * 64 + s * 16 + m) * DMODEL;

  v8f acc[4];
#pragma unroll
  for (int s = 0; s < 4; ++s) acc[s] = vzero8();

  for (int k0 = 0; k0 < DMODEL; k0 += 32) {
    __builtin_prefetch(arow + k0 + 256, 0, 1);
    v16bf a = frag_ld(arow + k0 + hf * 8, arow + k0 + 16 + hf * 8);
#pragma unroll
    for (int s = 0; s < 4; ++s) {
      v16bf b = frag_ld(brow[s] + k0 + hf * 8, brow[s] + k0 + 16 + hf * 8);
      acc[s] = wmma_bf16(a, b, acc[s]);
    }
  }

#pragma unroll
  for (int s = 0; s < 4; ++s) {
    const int e = nt * 64 + s * 16 + m;
    const float bias = bout[e];
#pragma unroll
    for (int r = 0; r < 8; ++r) {
      const int t = mt * 16 + hf * 8 + r;
      out[(size_t)t * DMODEL + e] = acc[s][r] + bias;
    }
  }
}

// ---------------- launcher ----------------
extern "C" void kernel_launch(void* const* d_in, const int* in_sizes, int n_in,
                              void* d_out, int out_size, void* d_ws, size_t ws_size,
                              hipStream_t stream) {
  const float* x    = (const float*)d_in[0];
  const float* Wqkv = (const float*)d_in[1];
  const float* bqkv = (const float*)d_in[2];
  const float* Wout = (const float*)d_in[3];
  const float* bout = (const float*)d_in[4];
  float* out = (float*)d_out;

  char* w = (char*)d_ws;
  __bf16* Xb  = (__bf16*)(w + 0);
  __bf16* WqT = (__bf16*)(w + 16777216);
  __bf16* WoT = (__bf16*)(w + 23068672);
  __bf16* Qs  = (__bf16*)(w + 25165824);
  __bf16* Ks  = (__bf16*)(w + 41943040);
  __bf16* Vt  = (__bf16*)(w + 58720256);
  __bf16* Ob  = (__bf16*)(w + 75497472);

  cvt_f32_bf16 <<<(BT * DMODEL)    / 256, 256, 0, stream>>>(x, Xb, BT * DMODEL);
  cvt_transpose<<<(DMODEL * NQKV)  / 256, 256, 0, stream>>>(Wqkv, WqT, DMODEL, NQKV);
  cvt_transpose<<<(DMODEL * DMODEL)/ 256, 256, 0, stream>>>(Wout, WoT, DMODEL, DMODEL);

  // 512 M-tiles x 48 N-tiles = 24576 waves / 8 per block
  qkv_gemm<<<3072, 256, 0, stream>>>(Xb, WqT, bqkv, Qs, Ks, Vt);
  // 64 (b,h) x 16 q-tile-blocks (8 q-tiles per block)
  attn_fwd<<<1024, 256, 0, stream>>>(Qs, Ks, Vt, Ob);
  // 512 M-tiles x 16 N-tiles = 8192 waves / 8 per block
  out_gemm<<<1024, 256, 0, stream>>>(Ob, WoT, bout, out);
}
